// KVCacheAttention_90529320665281
// MI455X (gfx1250) — matrix operand order repaired and checked by reading, not compiled, not run on hardware
//
#include <hip/hip_runtime.h>

typedef unsigned short u16;
typedef unsigned int   u32;
typedef unsigned long long u64;
typedef __attribute__((ext_vector_type(16))) __bf16 v16bf;
typedef __attribute__((ext_vector_type(8)))  float  v8f;
typedef __attribute__((ext_vector_type(4)))  unsigned int v4u;
typedef __attribute__((ext_vector_type(8)))  int v8i;
typedef __attribute__((ext_vector_type(4)))  int v4i;

#define H      2048
#define NH     16
#define HD     128
#define SEQ    2048
#define BATCH  2
#define MAXCTX 4096
#define CTX    4096
#define OFF    2048
#define MROWS  (BATCH*SEQ)   /* 4096 */
#define QKV_N  (3*H)         /* 6144 */

#if defined(__has_builtin)
#if __has_builtin(__builtin_amdgcn_tensor_load_to_lds) && __has_builtin(__builtin_amdgcn_s_wait_tensorcnt)
#define HAVE_TDM 1
#endif
#endif
#ifndef HAVE_TDM
#define HAVE_TDM 0
#endif

__device__ __forceinline__ u16 f2bf(float f) {
  u32 u = __float_as_uint(f);
  u32 r = (u + 0x7FFFu + ((u >> 16) & 1u)) >> 16;  // round-to-nearest-even
  return (u16)r;
}

union FragU { uint4 u[2]; v16bf v; };
union U4    { uint4 u; u16 s[8]; };

__device__ __forceinline__ v16bf ldfrag(const u16* p0, const u16* p1) {
  FragU f;
  f.u[0] = *(const uint4*)p0;
  f.u[1] = *(const uint4*)p1;
  return f.v;
}

// --- async global->LDS copy (GLOBAL_LOAD_ASYNC_TO_LDS_B128, GVS mode) -------
__device__ __forceinline__ void async_b128(void* lds_dst, const void* gbase, u32 voff) {
  u32 l = (u32)(size_t)lds_dst;  // low 32 bits of flat shared ptr = LDS offset
  asm volatile("global_load_async_to_lds_b128 %0, %1, %2"
               :: "v"(l), "v"(voff), "s"(gbase)
               : "memory");
}
__device__ __forceinline__ void wait_async() {
  asm volatile("s_wait_asynccnt 0x0" ::: "memory");
}

#if HAVE_TDM
// --- Tensor Data Mover: 2D bf16 tile -> LDS with per-row padding ------------
// D# per CDNA5 ISA 8.3/8.4. data_size=1 (2B). tensor_dim0 = row stride =
// tensor_dim0_stride (rows contiguous in a larger matrix), tile is
// tile_rows x tile_cols elements; LDS rows padded (pad encodings passed in).
__device__ __forceinline__ void tdm_load_2d(void* lds_dst, const void* gsrc,
                                            u32 tile_cols, u32 tile_rows,
                                            u32 stride_elems,
                                            u32 pad_interval, u32 pad_amount) {
  u64 ga = (u64)(size_t)gsrc;
  v4u g0 = {};
  g0[0] = 1u;                                             // count=1, user desc
  g0[1] = (u32)(size_t)lds_dst;                           // lds_addr (bytes)
  g0[2] = (u32)ga;                                        // global_addr[31:0]
  g0[3] = (u32)((ga >> 32) & 0x01FFFFFFu) | (2u << 30);   // addr[56:32], type=2
  v8i g1 = {};
  g1[0] = (int)((1u << 16)            // data_size = 2 bytes
              | (1u << 20)            // pad_enable
              | (pad_interval << 22)
              | (pad_amount << 25));
  g1[1] = (int)((stride_elems & 0xFFFFu) << 16);          // tensor_dim0 lo16
  g1[2] = (int)((stride_elems >> 16) | ((tile_rows & 0xFFFFu) << 16)); // dim0 hi / dim1 lo
  g1[3] = (int)(tile_cols << 16);                         // dim1 hi=0, tile_dim0
  g1[4] = (int)tile_rows;                                 // tile_dim1 (tile_dim2=0)
  g1[5] = (int)stride_elems;                              // tensor_dim0_stride lo32
  g1[6] = 0;
  g1[7] = 0;
  v4i z4 = {};
  v8i z8 = {};
  __builtin_amdgcn_tensor_load_to_lds(g0, g1, z4, z4, z8, 0);
}
#endif

// --- VALU (DPP) max-reduction across each 16-lane row -----------------------
template <int CTRL>
__device__ __forceinline__ float fmax_dpp(float x) {
  int y = __builtin_amdgcn_mov_dpp(__float_as_int(x), CTRL, 0xF, 0xF, true);
  return fmaxf(x, __int_as_float(y));
}
__device__ __forceinline__ float rowmax16(float x) {
  x = fmax_dpp<0xB1>(x);   // quad_perm(1,0,3,2)  : xor 1
  x = fmax_dpp<0x4E>(x);   // quad_perm(2,3,0,1)  : xor 2
  x = fmax_dpp<0x141>(x);  // row_half_mirror     : xor 7
  x = fmax_dpp<0x140>(x);  // row_mirror          : xor 15
  return x;
}

// ---------------------------------------------------------------- LayerNorm
__global__ __launch_bounds__(256) void ln_kernel(const float* __restrict__ x,
                                                 const float* __restrict__ g,
                                                 const float* __restrict__ bta,
                                                 u16* __restrict__ xb) {
  int row = blockIdx.x, tid = threadIdx.x;
  const float* xr = x + (size_t)row * H;
  float v[8]; float sum = 0.f, sq = 0.f;
#pragma unroll
  for (int e = 0; e < 8; e++) { int idx = e*256 + tid; float f = xr[idx]; v[e] = f; sum += f; sq += f*f; }
#pragma unroll
  for (int m = 16; m >= 1; m >>= 1) { sum += __shfl_xor(sum, m, 32); sq += __shfl_xor(sq, m, 32); }
  __shared__ float rs[8], rq[8];
  int wid = tid >> 5, lane = tid & 31;
  if (lane == 0) { rs[wid] = sum; rq[wid] = sq; }
  __syncthreads();
  float ts = 0.f, tq = 0.f;
#pragma unroll
  for (int i = 0; i < 8; i++) { ts += rs[i]; tq += rq[i]; }
  float mean = ts * (1.0f / H);
  float var  = tq * (1.0f / H) - mean * mean;
  float rstd = rsqrtf(var + 1e-5f);
#pragma unroll
  for (int e = 0; e < 8; e++) {
    int idx = e*256 + tid;
    float y = (v[e] - mean) * rstd * g[idx] + bta[idx];
    xb[(size_t)row * H + idx] = f2bf(y);
  }
}

// ------------------------------------------- fp32 [K][N] -> bf16 [N][K] (W^T)
__global__ void transpose_bf16(const float* __restrict__ in, u16* __restrict__ outp,
                               int K, int N) {
  __shared__ float tile[32][33];
  int tx = threadIdx.x, ty = threadIdx.y;
  int n0 = blockIdx.x * 32, k0 = blockIdx.y * 32;
#pragma unroll
  for (int j = 0; j < 4; j++)
    tile[ty + j*8][tx] = in[(size_t)(k0 + ty + j*8) * N + n0 + tx];
  __syncthreads();
#pragma unroll
  for (int j = 0; j < 4; j++)
    outp[(size_t)(n0 + ty + j*8) * K + k0 + tx] = f2bf(tile[tx][ty + j*8]);
}

// -------- cache fp32 (B,MAX,NH,HD) rows [0,OFF) -> bf16 [(b*NH+h)][t][d]
__global__ void conv_cache(const float* __restrict__ src, u16* __restrict__ dst) {
  int gid = blockIdx.x * 256 + threadIdx.x;  // one float4 per thread
  int d4  = gid & 31;
  int tmp = gid >> 5;
  int h = tmp & 15; tmp >>= 4;
  int t = tmp & 2047;
  int b = tmp >> 11;
  const float4 f = *(const float4*)(src + (((size_t)(b*MAXCTX + t))*NH + h)*HD + d4*4);
  size_t o = (((size_t)(b*NH + h))*CTX + t)*HD + d4*4;
  dst[o+0] = f2bf(f.x); dst[o+1] = f2bf(f.y); dst[o+2] = f2bf(f.z); dst[o+3] = f2bf(f.w);
}

// ---------------- tiled bf16 GEMM, TDM-staged, double-buffered (WMMA)
// MODE 0: QKV epilogue (scatter bf16 q/k/v).  MODE 1: fp32 out + bias.
template <int MODE>
__global__ __launch_bounds__(256) void gemm_bf16(
    const u16* __restrict__ A, const u16* __restrict__ Wt,
    const float* __restrict__ bias, float* __restrict__ Cout,
    u16* __restrict__ qb, u16* __restrict__ kb, u16* __restrict__ vb,
    int M, int N, int K) {
  __shared__ u16 abuf[2][128 * 40];
  __shared__ u16 bbuf[2][128 * 40];
  int tid = threadIdx.x;
  int wid = tid >> 5, lane = tid & 31;
  int half = lane >> 4, lr = lane & 15;
  int wm = wid & 1, wn = wid >> 1;
  int mb = blockIdx.y, nb = blockIdx.x;
  v8f zero = {};
  v8f acc[4][2];
#pragma unroll
  for (int i = 0; i < 4; i++)
#pragma unroll
    for (int j = 0; j < 2; j++) acc[i][j] = zero;

  const u16* Ab = A  + (size_t)(mb * 128) * K;
  const u16* Bb = Wt + (size_t)(nb * 128) * K;

  auto stage = [&](int bufi, int kk) {
#if HAVE_TDM
    // one TDM descriptor per 128x32 tile; LDS rows padded 64B+16B (pi=3,pa=3)
    if (wid == 0) tdm_load_2d(&abuf[bufi][0], Ab + kk, 32, 128, (u32)K, 3, 3);
    if (wid == 1) tdm_load_2d(&bbuf[bufi][0], Bb + kk, 32, 128, (u32)K, 3, 3);
#else
#pragma unroll
    for (int p = 0; p < 2; p++) {
      int c = tid + p * 256;
      int r = c >> 2, cx = c & 3;
      u32 voff = (u32)(((u32)r * K + kk + cx * 8) * 2);
      async_b128(&abuf[bufi][r*40 + cx*8], Ab, voff);
      async_b128(&bbuf[bufi][r*40 + cx*8], Bb, voff);
    }
#endif
  };
  auto stage_wait = [&]() {
#if HAVE_TDM
    if (wid < 2) __builtin_amdgcn_s_wait_tensorcnt(0);
#else
    wait_async();
#endif
    __syncthreads();
  };

  stage(0, 0);
  stage_wait();
  int cur = 0;

  for (int kk = 0; kk < K; kk += 32) {
    if (kk + 32 < K) stage(cur ^ 1, kk + 32);  // prefetch next tile via TDM

    v16bf af[4], bfr[2];
#pragma unroll
    for (int ti = 0; ti < 4; ti++) {
      int r = wm*64 + ti*16 + lr;
      af[ti] = ldfrag(&abuf[cur][r*40 + half*8], &abuf[cur][r*40 + half*8 + 16]);
    }
#pragma unroll
    for (int tj = 0; tj < 2; tj++) {
      int c = wn*32 + tj*16 + lr;
      bfr[tj] = ldfrag(&bbuf[cur][c*40 + half*8], &bbuf[cur][c*40 + half*8 + 16]);
    }
#pragma unroll
    for (int ti = 0; ti < 4; ti++)
#pragma unroll
      for (int tj = 0; tj < 2; tj++)
        acc[ti][tj] = __builtin_amdgcn_wmma_f32_16x16x32_bf16(
            false, af[ti], false, bfr[tj], (short)0, acc[ti][tj], false, false);

    stage_wait();
    cur ^= 1;
  }

#pragma unroll
  for (int ti = 0; ti < 4; ti++) {
#pragma unroll
    for (int tj = 0; tj < 2; tj++) {
      int n = nb*128 + wn*32 + tj*16 + lr;
      float bv = bias[n];
#pragma unroll
      for (int i = 0; i < 8; i++) {
        int m = mb*128 + wm*64 + ti*16 + half*8 + i;
        float val = acc[ti][tj][i] + bv;
        if (MODE == 1) {
          Cout[(size_t)m * N + n] = val;
        } else {
          int b = m >> 11, s = m & 2047;
          if (n < H) {
            int hh = n >> 7, d = n & 127;
            qb[(((size_t)(b*NH + hh))*SEQ + s)*HD + d] = f2bf(val * 0.08838834764831845f);
          } else if (n < 2*H) {
            int n2 = n - H; int hh = n2 >> 7, d = n2 & 127;
            kb[(((size_t)(b*NH + hh))*CTX + (OFF + s))*HD + d] = f2bf(val);
          } else {
            int n2 = n - 2*H; int hh = n2 >> 7, d = n2 & 127;
            vb[(((size_t)(b*NH + hh))*CTX + (OFF + s))*HD + d] = f2bf(val);
          }
        }
      }
    }
  }
}

// ------------------ flash attention, TDM K-tiles, double-buffered (WMMA)
__global__ __launch_bounds__(256) void attn_kernel(const u16* __restrict__ qb,
                                                   const u16* __restrict__ kb,
                                                   const u16* __restrict__ vb,
                                                   u16* __restrict__ ab) {
  __shared__ u16 kbuf[2][32 * 136];   // [key][d], padded (256B data + 16B pad)
  __shared__ u16 vbuf[2][128 * 40];   // [d][key], transposed, padded
  __shared__ u16 pbuf[8][16 * 40];    // per-wave P tiles
  int tid = threadIdx.x;
  int wid = tid >> 5, lane = tid & 31;
  int half = lane >> 4, lr = lane & 15;
  int bh = blockIdx.x;                // b*NH + h
  int qblk = blockIdx.y;
  int b = bh >> 4, h = bh & 15;
  int q0 = qblk*128 + wid*16;

  v16bf qf[4];
  const u16* qrow = qb + (((size_t)bh * SEQ) + q0 + lr) * HD;
#pragma unroll
  for (int dk = 0; dk < 4; dk++)
    qf[dk] = ldfrag(qrow + dk*32 + half*8, qrow + dk*32 + half*8 + 16);

  // all-ones B fragment: softmax denominator accumulates in a 9th WMMA column
  v16bf ones;
  {
    FragU f;
    f.u[0] = make_uint4(0x3F803F80u, 0x3F803F80u, 0x3F803F80u, 0x3F803F80u);
    f.u[1] = f.u[0];
    ones = f.v;
  }

  v8f zero = {};
  v8f oacc[8], lacc = zero;
#pragma unroll
  for (int j = 0; j < 8; j++) oacc[j] = zero;
  float mrow[8];
#pragma unroll
  for (int i = 0; i < 8; i++) mrow[i] = -1e30f;

  u16* pw = pbuf[wid];
  const u16* kbase = kb + (size_t)bh * CTX * HD;
  const u16* vbase = vb + (size_t)bh * CTX * HD;

  auto stage = [&](int bufi, int t0) {
#if HAVE_TDM
    // 32x128 bf16 K-tile, LDS rows padded 256B+16B (pi=5, pa=3)
    if (wid == 0) tdm_load_2d(&kbuf[bufi][0], kbase + (size_t)t0 * HD, 128, 32, 128, 5, 3);
#else
#pragma unroll
    for (int p = 0; p < 2; p++) {
      int c = tid + p * 256;
      int r = c >> 4, cx = c & 15;
      async_b128(&kbuf[bufi][r*136 + cx*8], kbase, (u32)((((u32)t0 + r)*HD + cx*8) * 2));
    }
#endif
    {                                   // V tile, transposed into [d][key]
      int key = tid >> 3, dseg = tid & 7;
      const u16* src = vbase + (size_t)(t0 + key)*HD + dseg*16;
      U4 a0, a1; a0.u = *(const uint4*)src; a1.u = *(const uint4*)(src + 8);
#pragma unroll
      for (int e = 0; e < 8; e++) vbuf[bufi][(dseg*16 + e)*40 + key]     = a0.s[e];
#pragma unroll
      for (int e = 0; e < 8; e++) vbuf[bufi][(dseg*16 + 8 + e)*40 + key] = a1.s[e];
    }
  };
  auto stage_wait = [&]() {
#if HAVE_TDM
    if (wid == 0) __builtin_amdgcn_s_wait_tensorcnt(0);
#else
    wait_async();
#endif
    __syncthreads();
  };

  stage(0, 0);
  stage_wait();
  int cur = 0;

  for (int t0 = 0; t0 < CTX; t0 += 32) {
    if (t0 + 32 < CTX) stage(cur ^ 1, t0 + 32);  // prefetch next K/V tiles

    v8f sc[2]; sc[0] = zero; sc[1] = zero;
#pragma unroll
    for (int kt = 0; kt < 2; kt++) {
      int key = kt*16 + lr;
#pragma unroll
      for (int dk = 0; dk < 4; dk++) {
        v16bf kf = ldfrag(&kbuf[cur][key*136 + dk*32 + half*8],
                          &kbuf[cur][key*136 + dk*32 + half*8 + 16]);
        sc[kt] = __builtin_amdgcn_wmma_f32_16x16x32_bf16(
            false, qf[dk], false, kf, (short)0, sc[kt], false, false);
      }
    }

    float ps[2][8];
#pragma unroll
    for (int i = 0; i < 8; i++) {
      float cm = rowmax16(fmaxf(sc[0][i], sc[1][i]));   // DPP, no LDS traffic
      float mnew  = fmaxf(mrow[i], cm);
      float scale = __expf(mrow[i] - mnew);
      ps[0][i] = __expf(sc[0][i] - mnew);
      ps[1][i] = __expf(sc[1][i] - mnew);
      mrow[i] = mnew;
#pragma unroll
      for (int j = 0; j < 8; j++) oacc[j][i] *= scale;
      lacc[i] *= scale;
    }
#pragma unroll
    for (int kt = 0; kt < 2; kt++)
#pragma unroll
      for (int i = 0; i < 8; i++)
        pw[(half*8 + i)*40 + kt*16 + lr] = f2bf(ps[kt][i]);
    __builtin_amdgcn_wave_barrier();    // wave-local LDS RAW; DS pipe is in-order

    v16bf pf = ldfrag(&pw[lr*40 + half*8], &pw[lr*40 + half*8 + 16]);
#pragma unroll
    for (int j = 0; j < 8; j++) {
      int d = j*16 + lr;
      v16bf vf = ldfrag(&vbuf[cur][d*40 + half*8], &vbuf[cur][d*40 + half*8 + 16]);
      oacc[j] = __builtin_amdgcn_wmma_f32_16x16x32_bf16(
          false, pf, false, vf, (short)0, oacc[j], false, false);
    }
    lacc = __builtin_amdgcn_wmma_f32_16x16x32_bf16(
        false, pf, false, ones, (short)0, lacc, false, false);

    stage_wait();
    cur ^= 1;
  }

  float inv[8];
#pragma unroll
  for (int i = 0; i < 8; i++) inv[i] = 1.0f / lacc[i];
#pragma unroll
  for (int j = 0; j < 8; j++)
#pragma unroll
    for (int i = 0; i < 8; i++) {
      int srow = q0 + half*8 + i;
      size_t o = ((size_t)(b*SEQ + srow)) * H + h*HD + j*16 + lr;
      ab[o] = f2bf(oacc[j][i] * inv[i]);
    }
}

// ----------------------------------------------------------------- launch
extern "C" void kernel_launch(void* const* d_in, const int* in_sizes, int n_in,
                              void* d_out, int out_size, void* d_ws, size_t ws_size,
                              hipStream_t stream) {
  (void)in_sizes; (void)n_in; (void)out_size; (void)ws_size;
  const float* tokens  = (const float*)d_in[0];
  const float* cache_k = (const float*)d_in[1];
  const float* cache_v = (const float*)d_in[2];
  const float* gamma   = (const float*)d_in[3];
  const float* beta    = (const float*)d_in[4];
  const float* qkv_w   = (const float*)d_in[5];
  const float* qkv_b   = (const float*)d_in[6];
  const float* proj_w  = (const float*)d_in[7];
  const float* proj_b  = (const float*)d_in[8];
  float* out = (float*)d_out;

  char* w = (char*)d_ws;
  u16* xb    = (u16*)w; w += (size_t)MROWS * H * 2;          // 16 MB
  u16* wqkv  = (u16*)w; w += (size_t)QKV_N * H * 2;          // 24 MB
  u16* wproj = (u16*)w; w += (size_t)H * H * 2;              //  8 MB
  u16* qb    = (u16*)w; w += (size_t)BATCH*NH*SEQ*HD * 2;    // 16 MB
  u16* kb    = (u16*)w; w += (size_t)BATCH*NH*CTX*HD * 2;    // 32 MB
  u16* vb    = (u16*)w; w += (size_t)BATCH*NH*CTX*HD * 2;    // 32 MB
  u16* ab    = (u16*)w; w += (size_t)MROWS * H * 2;          // 16 MB

  ln_kernel<<<MROWS, 256, 0, stream>>>(tokens, gamma, beta, xb);
  transpose_bf16<<<dim3(QKV_N/32, H/32), dim3(32, 8), 0, stream>>>(qkv_w, wqkv, H, QKV_N);
  transpose_bf16<<<dim3(H/32,     H/32), dim3(32, 8), 0, stream>>>(proj_w, wproj, H, H);
  int cvt_blocks = (BATCH*NH*OFF*HD/4) / 256;                // 8192
  conv_cache<<<cvt_blocks, 256, 0, stream>>>(cache_k, kb);
  conv_cache<<<cvt_blocks, 256, 0, stream>>>(cache_v, vb);
  gemm_bf16<0><<<dim3(QKV_N/128, MROWS/128), 256, 0, stream>>>(
      xb, wqkv, qkv_b, nullptr, qb, kb, vb, MROWS, QKV_N, H);
  attn_kernel<<<dim3(BATCH*NH, SEQ/128), 256, 0, stream>>>(qb, kb, vb, ab);
  gemm_bf16<1><<<dim3(H/128, MROWS/128), 256, 0, stream>>>(
      ab, wproj, proj_b, out, nullptr, nullptr, nullptr, MROWS, H, H);
}
